// Combineall_49134425866943
// MI455X (gfx1250) — matrix operand order repaired and checked by reading, not compiled, not run on hardware
//
#include <hip/hip_runtime.h>
#include <math.h>

typedef __attribute__((ext_vector_type(16))) _Float16 v16h;
typedef __attribute__((ext_vector_type(8)))  float    v8f;

#define EPSV 1e-5f

// ---------------------------------------------------------------------------
// K0: start offsets per graph via binary search (batch arrays are sorted).
// ---------------------------------------------------------------------------
__global__ void k_bounds(const int* __restrict__ b1, const int* __restrict__ b2,
                         int N, int B, int* __restrict__ start1, int* __restrict__ start2) {
    int t = blockIdx.x * blockDim.x + threadIdx.x;
    if (t >= 2 * (B + 1)) return;
    int which = (t >= (B + 1));
    int g = which ? (t - (B + 1)) : t;
    const int* b = which ? b2 : b1;
    int* s = which ? start2 : start1;
    int lo = 0, hi = N;               // lower_bound: first idx with b[idx] >= g
    while (lo < hi) { int mid = (lo + hi) >> 1; if (b[mid] < g) lo = mid + 1; else hi = mid; }
    s[g] = lo;
}

// ---------------------------------------------------------------------------
// K1: per-graph per-feature sum, sum-of-squares, mean. One block per (graph, input).
// 256 threads = 4 row-subsets x 64 float4 lanes: wide b128 loads, rows split
// 4-ways for parallelism, partials combined once through LDS. Requires F==256.
// ---------------------------------------------------------------------------
__global__ void k_graph_stats(const float* __restrict__ x1, const float* __restrict__ x2,
                              const int* __restrict__ start1, const int* __restrict__ start2, int F,
                              float* __restrict__ gsum1, float* __restrict__ gsq1, float* __restrict__ mean1,
                              float* __restrict__ gsum2, float* __restrict__ gsq2, float* __restrict__ mean2) {
    __shared__ float acc[2][4][256];
    int g = blockIdx.x;
    int which = blockIdx.y;
    const float* x = which ? x2 : x1;
    const int* st  = which ? start2 : start1;
    float* gsum = which ? gsum2 : gsum1;
    float* gsq  = which ? gsq2  : gsq1;
    float* mean = which ? mean2 : mean1;
    int base = st[g];
    int n = st[g + 1] - base;

    int t   = threadIdx.x;
    int sub = t >> 6;          // 0..3: row subset
    int fq  = t & 63;          // float4 column
    const float4* x4 = (const float4*)x;
    int F4 = F >> 2;           // 64

    float s0 = 0.f, s1 = 0.f, s2 = 0.f, s3 = 0.f;
    float q0 = 0.f, q1 = 0.f, q2 = 0.f, q3 = 0.f;
    for (int p = sub; p < n; p += 4) {
        float4 v = x4[(size_t)(base + p) * F4 + fq];
        s0 += v.x; s1 += v.y; s2 += v.z; s3 += v.w;
        q0 += v.x * v.x; q1 += v.y * v.y; q2 += v.z * v.z; q3 += v.w * v.w;
    }
    int fb = fq * 4;
    acc[0][sub][fb + 0] = s0; acc[0][sub][fb + 1] = s1;
    acc[0][sub][fb + 2] = s2; acc[0][sub][fb + 3] = s3;
    acc[1][sub][fb + 0] = q0; acc[1][sub][fb + 1] = q1;
    acc[1][sub][fb + 2] = q2; acc[1][sub][fb + 3] = q3;
    __syncthreads();

    int f = t;                 // 0..255
    float S = acc[0][0][f] + acc[0][1][f] + acc[0][2][f] + acc[0][3][f];
    float Q = acc[1][0][f] + acc[1][1][f] + acc[1][2][f] + acc[1][3][f];
    gsum[(size_t)g * F + f] = S;
    gsq [(size_t)g * F + f] = Q;
    mean[(size_t)g * F + f] = (n > 0) ? S / (float)n : 0.f;
}

// ---------------------------------------------------------------------------
// K2: batchnorm stats over the dense padded tensor (pads included): m,
// scale=gamma/sqrt(v+eps), pad value P = tanh((0-m)*scale + beta).
// ---------------------------------------------------------------------------
__global__ void k_feat_stats(const float* __restrict__ gsum1, const float* __restrict__ gsq1,
                             const float* __restrict__ gsum2, const float* __restrict__ gsq2,
                             const float* __restrict__ gamma, const float* __restrict__ beta,
                             int B, int F, const int* __restrict__ nmaxp,
                             float* __restrict__ m1, float* __restrict__ s1, float* __restrict__ P1,
                             float* __restrict__ m2, float* __restrict__ s2, float* __restrict__ P2) {
    int which = blockIdx.x;
    int f = threadIdx.x;
    const float* gs = which ? gsum2 : gsum1;
    const float* gq = which ? gsq2  : gsq1;
    float fs = 0.f, fq = 0.f;
    for (int g = 0; g < B; ++g) {
        fs += gs[(size_t)g * F + f];
        fq += gq[(size_t)g * F + f];
    }
    float denom = (float)B * (float)nmaxp[0];
    float m   = fs / denom;
    float var = fq / denom - m * m;
    float s   = gamma[f] * rsqrtf(var + EPSV);
    float P   = tanhf((0.f - m) * s + beta[f]);
    if (which) { m2[f] = m; s2[f] = s; P2[f] = P; }
    else       { m1[f] = m; s1[f] = s; P1[f] = P; }
}

// ---------------------------------------------------------------------------
// K3: tg = tanh(mean @ W) via V_WMMA_F32_16X16X32_F16, one wave per 16x16 tile.
// Fragment layouts per CDNA5 ISA 7.12.2 (wave32):
//   A (16x32 f16): lane L: M=L&15; half=L>>4; elem e -> K=(e<8?0:16)+half*8+(e&7)
//   B (32x16 f16): lane L: N=L&15; half=L>>4; elem e -> K=(e<8?0:16)+half*8+(e&7)
//   D (16x16 f32): vgpr r: M=(L<16 ? r : r+8); N=L&15
// F==256 path fully unrolled -> 8 WMMAs, loads/cvts scheduled across them.
// ---------------------------------------------------------------------------
__device__ __forceinline__ v8f wmma_step(const float* __restrict__ A, const float* __restrict__ W,
                                         int arow, int tileN, int F, int k0,
                                         int hs, int r15, v8f c) {
    v16h a, b;
#pragma unroll
    for (int e = 0; e < 16; ++e) {
        int ka = (e < 8 ? 0 : 16) + hs * 8 + (e & 7);
        a[e] = (_Float16)A[(size_t)arow * F + k0 + ka];
        int kb = (e < 8 ? 0 : 16) + hs * 8 + (e & 7);
        b[e] = (_Float16)W[(size_t)(k0 + kb) * F + tileN + r15];
    }
    return __builtin_amdgcn_wmma_f32_16x16x32_f16(false, a, false, b, (short)0, c, false, false);
}

__global__ void k_gemm_tanh(const float* __restrict__ mean1, const float* __restrict__ mean2,
                            const float* __restrict__ W, int F, int Btot,
                            float* __restrict__ tg1, float* __restrict__ tg2) {
    int tileM = blockIdx.x * 16;
    int tileN = blockIdx.y * 16;
    int which = blockIdx.z;
    const float* A = which ? mean2 : mean1;
    float* D       = which ? tg2   : tg1;

    int lane = threadIdx.x;        // 0..31, wave32
    int hs   = lane >> 4;
    int r15  = lane & 15;

    int arow = tileM + r15;
    if (arow >= Btot) arow = Btot - 1;   // clamp (stores are guarded)

    v8f c = {};
    if (F == 256) {
#pragma unroll
        for (int kk = 0; kk < 8; ++kk)
            c = wmma_step(A, W, arow, tileN, 256, kk * 32, hs, r15, c);
    } else {
        for (int k0 = 0; k0 + 32 <= F; k0 += 32)
            c = wmma_step(A, W, arow, tileN, F, k0, hs, r15, c);
    }
#pragma unroll
    for (int r = 0; r < 8; ++r) {
        int row = hs * 8 + r;
        if (tileM + row < Btot)
            D[(size_t)(tileM + row) * F + tileN + r15] = tanhf(c[r]);
    }
}

// ---------------------------------------------------------------------------
// Wave-level (32-lane) all-reduce sum via shfl_xor (no LDS, no syncs).
// ---------------------------------------------------------------------------
__device__ __forceinline__ float wave_allreduce(float v) {
#pragma unroll
    for (int off = 16; off > 0; off >>= 1) v += __shfl_xor(v, off, 32);
    return v;
}

// ---------------------------------------------------------------------------
// K4: fused per-graph pass (pass 2 over x1,x2). One block (8 waves) per graph.
// Wave wid owns positions p = wid, wid+8, ... -> no block-wide syncs in the hot
// loop. Lane holds 8 CONTIGUOUS features f = lane*8 + k -> each row is two
// global_load_b128 per lane (wave covers all 256 features, 1 KB coalesced).
// Gate dot = 8 FMAs + 5-step shfl_xor wave all-reduce.
//  - coef_n = sigmoid(<x_n, tg[g]>) ; e += coef_n * x_n
//  - scoreh[g,f] = sum_p tanh(bn1(h1)) * tanh(bn2(h2)) over the dense layout,
//    pad tail (nmax - max(n1,n2)) * P1f * P2f added by wave 0.
// ---------------------------------------------------------------------------
__global__ void k_main(const float* __restrict__ x1, const float* __restrict__ x2,
                       const int* __restrict__ start1, const int* __restrict__ start2,
                       const float* __restrict__ tg1, const float* __restrict__ tg2,
                       const float* __restrict__ m1, const float* __restrict__ s1, const float* __restrict__ P1,
                       const float* __restrict__ m2, const float* __restrict__ s2, const float* __restrict__ P2,
                       const float* __restrict__ beta, const int* __restrict__ nmaxp,
                       int F, int OUTW,
                       float* __restrict__ e1, float* __restrict__ e2, float* __restrict__ out) {
    __shared__ float part[3][8][256];      // e1 / e2 / scoreh partials per wave
    int g    = blockIdx.x;
    int tid  = threadIdx.x;
    int lane = tid & 31;
    int wid  = tid >> 5;

    int b1 = start1[g], n1 = start1[g + 1] - b1;
    int b2 = start2[g], n2 = start2[g + 1] - b2;
    int nmax = nmaxp[0];

    // per-lane feature registers: f = lane*8 + k (contiguous per lane)
    float tg1v[8], tg2v[8], m1v[8], s1v[8], p1v[8], m2v[8], s2v[8], p2v[8], btv[8];
#pragma unroll
    for (int k = 0; k < 8; ++k) {
        int f = lane * 8 + k;
        tg1v[k] = tg1[(size_t)g * F + f];
        tg2v[k] = tg2[(size_t)g * F + f];
        m1v[k] = m1[f]; s1v[k] = s1[f]; p1v[k] = P1[f];
        m2v[k] = m2[f]; s2v[k] = s2[f]; p2v[k] = P2[f];
        btv[k] = beta[f];
    }

    float a1[8], a2[8], sc[8];
#pragma unroll
    for (int k = 0; k < 8; ++k) { a1[k] = 0.f; a2[k] = 0.f; sc[k] = 0.f; }

    int F4 = F >> 2;                       // float4 stride per row
    const float4* x14 = (const float4*)x1;
    const float4* x24 = (const float4*)x2;

    int pmax = n1 > n2 ? n1 : n2;
    for (int p = wid; p < pmax; p += 8) {      // uniform within each wave
        float t1a[8], t2a[8];
        if (p < n1) {
            float v[8];
            const float4* row = x14 + (size_t)(b1 + p) * F4 + lane * 2;
            float4 u0 = row[0], u1 = row[1];   // 2x global_load_b128
            v[0] = u0.x; v[1] = u0.y; v[2] = u0.z; v[3] = u0.w;
            v[4] = u1.x; v[5] = u1.y; v[6] = u1.z; v[7] = u1.w;
            float d = 0.f;
#pragma unroll
            for (int k = 0; k < 8; ++k) d += v[k] * tg1v[k];
            d = wave_allreduce(d);
            float coef = 1.f / (1.f + expf(-d));
#pragma unroll
            for (int k = 0; k < 8; ++k) {
                a1[k] += coef * v[k];
                t1a[k] = tanhf((v[k] - m1v[k]) * s1v[k] + btv[k]);
            }
        } else {
#pragma unroll
            for (int k = 0; k < 8; ++k) t1a[k] = p1v[k];
        }
        if (p < n2) {
            float v[8];
            const float4* row = x24 + (size_t)(b2 + p) * F4 + lane * 2;
            float4 u0 = row[0], u1 = row[1];
            v[0] = u0.x; v[1] = u0.y; v[2] = u0.z; v[3] = u0.w;
            v[4] = u1.x; v[5] = u1.y; v[6] = u1.z; v[7] = u1.w;
            float d = 0.f;
#pragma unroll
            for (int k = 0; k < 8; ++k) d += v[k] * tg2v[k];
            d = wave_allreduce(d);
            float coef = 1.f / (1.f + expf(-d));
#pragma unroll
            for (int k = 0; k < 8; ++k) {
                a2[k] += coef * v[k];
                t2a[k] = tanhf((v[k] - m2v[k]) * s2v[k] + btv[k]);
            }
        } else {
#pragma unroll
            for (int k = 0; k < 8; ++k) t2a[k] = p2v[k];
        }
#pragma unroll
        for (int k = 0; k < 8; ++k) sc[k] += t1a[k] * t2a[k];
    }
    if (wid == 0) {                            // closed-form pad tail, once
        float tail = (float)(nmax - pmax);
#pragma unroll
        for (int k = 0; k < 8; ++k) sc[k] += tail * p1v[k] * p2v[k];
    }

    // combine wave partials once through LDS
#pragma unroll
    for (int k = 0; k < 8; ++k) {
        int f = lane * 8 + k;
        part[0][wid][f] = a1[k];
        part[1][wid][f] = a2[k];
        part[2][wid][f] = sc[k];
    }
    __syncthreads();
    int f = tid;                                // 0..255
    float A1 = 0.f, A2 = 0.f, SC = 0.f;
#pragma unroll
    for (int w = 0; w < 8; ++w) {
        A1 += part[0][w][f];
        A2 += part[1][w][f];
        SC += part[2][w][f];
    }
    e1[(size_t)g * F + f] = A1;
    e2[(size_t)g * F + f] = A2;
    out[(size_t)g * OUTW + 48 + f] = SC;       // scoreh occupies cols 48..48+F-1
}

// ---------------------------------------------------------------------------
// 256-thread block reduction (used only by the tiny conv head).
// ---------------------------------------------------------------------------
__device__ __forceinline__ float block_reduce_256(float v, float* red) {
#pragma unroll
    for (int off = 16; off > 0; off >>= 1) v += __shfl_down(v, off, 32);
    int wid = threadIdx.x >> 5;
    if ((threadIdx.x & 31) == 0) red[wid] = v;
    __syncthreads();
    float t = 0.f;
#pragma unroll
    for (int w = 0; w < 8; ++w) t += red[w];
    __syncthreads();
    return t;
}

// ---------------------------------------------------------------------------
// K5: conv similarity head. One block per (graph, kernel/stride combo).
// ---------------------------------------------------------------------------
__global__ void k_conv(const float* __restrict__ e1, const float* __restrict__ e2,
                       const float* __restrict__ w1, const float* __restrict__ w2,
                       const float* __restrict__ w3, const float* __restrict__ w4,
                       const float* __restrict__ w5, const float* __restrict__ w6,
                       const float* __restrict__ w7, const float* __restrict__ w8,
                       int F, int OUTW, float* __restrict__ out) {
    __shared__ float se1[256];
    __shared__ float se2[256];
    __shared__ float red[8];
    int g = blockIdx.x;
    int pair = blockIdx.y;
    int ki = pair / 3;        // width ki+1
    int si = pair % 3;        // stride si+1
    int K = ki + 1, S = si + 1;
    const float* wk;
    switch (ki) {
        case 0: wk = w1; break; case 1: wk = w2; break;
        case 2: wk = w3; break; case 3: wk = w4; break;
        case 4: wk = w5; break; case 5: wk = w6; break;
        case 6: wk = w7; break; default: wk = w8; break;
    }
    const float* w = wk + (size_t)si * 4 * K;   // (4, 1, K) slice

    int t = threadIdx.x;
    se1[t] = e1[(size_t)g * F + t];
    se2[t] = e2[(size_t)g * F + t];
    __syncthreads();

    int L = (F - K) / S + 1;
    float hloc = 0.f;
    float dloc[4] = {0.f, 0.f, 0.f, 0.f};
    if (t < L) {
        int j0 = t * S;
#pragma unroll
        for (int o = 0; o < 4; ++o) {
            float c1 = 0.f, c2 = 0.f;
            for (int tt = 0; tt < K; ++tt) {
                float wv = w[o * K + tt];
                c1 += wv * se1[j0 + tt];
                c2 += wv * se2[j0 + tt];
            }
            hloc += tanhf(c1) * tanhf(c2);
            float d = c1 - c2;
            dloc[o] = d * d;
        }
    }
    float hs = block_reduce_256(hloc, red);
    float cs = 0.f;
#pragma unroll
    for (int o = 0; o < 4; ++o) {
        float Do = block_reduce_256(dloc[o], red);
        cs += expf(-Do * 0.25f);
    }
    if (t == 0) {
        out[(size_t)g * OUTW + pair * 2 + 0] = hs / (4.f * (float)L);
        out[(size_t)g * OUTW + pair * 2 + 1] = cs * 0.25f;
    }
}

// ---------------------------------------------------------------------------
// Host launcher
// ---------------------------------------------------------------------------
extern "C" void kernel_launch(void* const* d_in, const int* in_sizes, int n_in,
                              void* d_out, int out_size, void* d_ws, size_t ws_size,
                              hipStream_t stream) {
    const float* x1    = (const float*)d_in[0];
    const float* x2    = (const float*)d_in[1];
    const float* W     = (const float*)d_in[2];
    const float* gamma = (const float*)d_in[3];
    const float* beta  = (const float*)d_in[4];
    const float* w1 = (const float*)d_in[5];
    const float* w2 = (const float*)d_in[6];
    const float* w3 = (const float*)d_in[7];
    const float* w4 = (const float*)d_in[8];
    const float* w5 = (const float*)d_in[9];
    const float* w6 = (const float*)d_in[10];
    const float* w7 = (const float*)d_in[11];
    const float* w8 = (const float*)d_in[12];
    const int* batch1 = (const int*)d_in[13];
    const int* batch2 = (const int*)d_in[14];
    const int* nmaxp  = (const int*)d_in[16];

    const int F = in_sizes[3];          // 256
    const int N = in_sizes[13];         // nodes
    const int OUTW = F + 48;            // 24 combos * 2 + F
    const int B = out_size / OUTW;      // 512

    float* out = (float*)d_out;

    // workspace carve-out
    char* ws = (char*)d_ws;
    size_t off = 0;
    auto carve = [&](size_t bytes) -> char* {
        char* p = ws + off;
        off += (bytes + 255) & ~(size_t)255;
        return p;
    };
    int*   start1 = (int*)  carve((size_t)(B + 1) * sizeof(int));
    int*   start2 = (int*)  carve((size_t)(B + 1) * sizeof(int));
    float* gsum1  = (float*)carve((size_t)B * F * sizeof(float));
    float* gsq1   = (float*)carve((size_t)B * F * sizeof(float));
    float* gsum2  = (float*)carve((size_t)B * F * sizeof(float));
    float* gsq2   = (float*)carve((size_t)B * F * sizeof(float));
    float* mean1  = (float*)carve((size_t)B * F * sizeof(float));
    float* mean2  = (float*)carve((size_t)B * F * sizeof(float));
    float* tg1    = (float*)carve((size_t)B * F * sizeof(float));
    float* tg2    = (float*)carve((size_t)B * F * sizeof(float));
    float* e1     = (float*)carve((size_t)B * F * sizeof(float));
    float* e2     = (float*)carve((size_t)B * F * sizeof(float));
    float* m1 = (float*)carve((size_t)F * sizeof(float));
    float* s1 = (float*)carve((size_t)F * sizeof(float));
    float* P1 = (float*)carve((size_t)F * sizeof(float));
    float* m2 = (float*)carve((size_t)F * sizeof(float));
    float* s2 = (float*)carve((size_t)F * sizeof(float));
    float* P2 = (float*)carve((size_t)F * sizeof(float));
    (void)ws_size; (void)n_in;

    // K0: segment boundaries
    {
        int tot = 2 * (B + 1);
        k_bounds<<<dim3((tot + 255) / 256), dim3(256), 0, stream>>>(batch1, batch2, N, B, start1, start2);
    }
    // K1: per-graph sums / sumsq / mean  (pass 1 over x1,x2, b128 loads)
    k_graph_stats<<<dim3(B, 2), dim3(256), 0, stream>>>(x1, x2, start1, start2, F,
                                                        gsum1, gsq1, mean1, gsum2, gsq2, mean2);
    // K2: global batchnorm stats + pad values
    k_feat_stats<<<dim3(2), dim3(F), 0, stream>>>(gsum1, gsq1, gsum2, gsq2, gamma, beta,
                                                  B, F, nmaxp, m1, s1, P1, m2, s2, P2);
    // K3: tg = tanh(mean @ W) with WMMA (one wave per 16x16 tile, 8 unrolled WMMAs)
    k_gemm_tanh<<<dim3((B + 15) / 16, F / 16, 2), dim3(32), 0, stream>>>(mean1, mean2, W, F, B, tg1, tg2);
    // K4: fused gates + readout sums + scoreh  (pass 2 over x1,x2, wave-per-node, b128 loads)
    k_main<<<dim3(B), dim3(256), 0, stream>>>(x1, x2, start1, start2, tg1, tg2,
                                              m1, s1, P1, m2, s2, P2, beta, nmaxp,
                                              F, OUTW, e1, e2, out);
    // K5: conv similarity head
    k_conv<<<dim3(B, 24), dim3(256), 0, stream>>>(e1, e2, w1, w2, w3, w4, w5, w6, w7, w8,
                                                  F, OUTW, out);
}
